// OFFChannelGINNode_57672820851274
// MI455X (gfx1250) — matrix operand order repaired and checked by reading, not compiled, not run on hardware
//
#include <hip/hip_runtime.h>
#include <hip/hip_fp16.h>

// ---------------------------------------------------------------------------
// GIN 2-layer GNN for MI455X (gfx1250, wave32).
// Edge scatter phases use native global_atomic_add_f32; dense node GEMMs use
// v_wmma_f32_16x16x32_f16 (f16 in, f32 accumulate). Weights are staged into
// LDS pre-swizzled into WMMA B-fragment order so each fragment is one
// contiguous 32-byte LDS read (2x ds_load_b128) instead of 16 scalar loads.
// ---------------------------------------------------------------------------

typedef __attribute__((ext_vector_type(16))) _Float16 v16h;
typedef __attribute__((ext_vector_type(8)))  float    v8f;

__device__ __forceinline__ float gelu_exact(float x) {
  return 0.5f * x * (1.0f + erff(x * 0.70710678118654752f));
}

__device__ __forceinline__ void atomicAddF(float* p, float v) {
  // Lowers to global_atomic_add_f32 (non-returning) on gfx1250.
  __hip_atomic_fetch_add(p, v, __ATOMIC_RELAXED, __HIP_MEMORY_SCOPE_AGENT);
}

// --- WMMA fragment builders (CDNA5 ISA 7.12.2 layouts, wave32) --------------
// A (16xK f16, K-chunk of 32 starting at kt*32):
//   lane l: row = l&15;  VGPR v holds K = ((v<4)?0:16) + (l>>4)*8 + (v&3)*2 + {0,1}
//   -> per lane this is two contiguous 8-half (16 B) chunks; compiler merges
//      the element loads into ds_load_b128 pairs.
__device__ __forceinline__ v16h load_a_frag(const _Float16* tile, int lane,
                                            int kt, int rowStride) {
  const _Float16* rp = tile + (size_t)(lane & 15) * rowStride + kt * 32;
  int kg = (lane >> 4) * 8;
  v16h a;
#pragma unroll
  for (int v = 0; v < 8; ++v) {
    int kb = ((v < 4) ? 0 : 16) + kg + (v & 3) * 2;
    a[2 * v]     = rp[kb];
    a[2 * v + 1] = rp[kb + 1];
  }
  return a;
}

// B fragments are pre-swizzled in LDS: fragment (kt, nt) for lane l lives at
// ((kt*NT + nt)*32 + l)*16 .. +15, i.e. one aligned 32-byte contiguous read.
__device__ __forceinline__ v16h load_b_frag_sw(const _Float16* sW, int lane,
                                               int kt, int NT, int nt) {
  return *(const v16h*)(sW + (((size_t)(kt * NT + nt) * 32 + lane) << 4));
}

// Stage a row-major [K x NCOLS] f32 weight matrix into LDS as f16 in
// B-fragment-swizzled order.  Inverse map of load_b_frag_sw:
//   elem = i & 15; frag = i >> 4; lane = frag & 31; tile = frag >> 5;
//   nt = tile % NT; kt = tile / NT;
//   k = kt*32 + ((elem>>3)*16) + (lane>>4)*8 + (elem&7);  col = (lane&15) + nt*16
template <int K, int NCOLS>
__device__ __forceinline__ void stage_w_swizzled(const float* __restrict__ W,
                                                 _Float16* __restrict__ sW) {
  constexpr int NT = NCOLS / 16;
  for (int i = threadIdx.x; i < K * NCOLS; i += blockDim.x) {
    int elem = i & 15;
    int frag = i >> 4;
    int lane = frag & 31;
    int tile = frag >> 5;
    int nt = tile % NT;
    int kt = tile / NT;
    int k = kt * 32 + ((elem >> 3) * 16) + ((lane >> 4) * 8) + (elem & 7);
    int col = (lane & 15) + nt * 16;
    sW[i] = (_Float16)W[k * NCOLS + col];
  }
}

// ---------------------------------------------------------------------------
// Kernel 1: per-edge MLP (3->16 GELU ->16) + scatter concat(nodes[src], e_emb)
// into agg1[dst] (21 channels) with f32 atomics.
// ---------------------------------------------------------------------------
__global__ void k_edge1(const float* __restrict__ nodes, const int* __restrict__ eidx,
                        const float* __restrict__ efeat,
                        const float* __restrict__ We1, const float* __restrict__ be1,
                        const float* __restrict__ We2, const float* __restrict__ be2,
                        float* __restrict__ agg1, int E) {
  __shared__ float sWe1[48], sbe1[16], sWe2[256], sbe2[16];
  for (int i = threadIdx.x; i < 48; i += blockDim.x) sWe1[i] = We1[i];
  for (int i = threadIdx.x; i < 16; i += blockDim.x) { sbe1[i] = be1[i]; sbe2[i] = be2[i]; }
  for (int i = threadIdx.x; i < 256; i += blockDim.x) sWe2[i] = We2[i];
  __syncthreads();
  int e = blockIdx.x * blockDim.x + threadIdx.x;
  if (e >= E) return;
  float f0 = efeat[3 * (size_t)e], f1 = efeat[3 * (size_t)e + 1], f2 = efeat[3 * (size_t)e + 2];
  float t[16];
#pragma unroll
  for (int j = 0; j < 16; ++j)
    t[j] = gelu_exact(f0 * sWe1[j] + f1 * sWe1[16 + j] + f2 * sWe1[32 + j] + sbe1[j]);
  float emb[16];
#pragma unroll
  for (int j = 0; j < 16; ++j) {
    float a = sbe2[j];
#pragma unroll
    for (int k = 0; k < 16; ++k) a += t[k] * sWe2[k * 16 + j];
    emb[j] = a;
  }
  int s = eidx[e], d = eidx[(size_t)E + e];
  const float* np = nodes + (size_t)s * 5;
  float* op = agg1 + (size_t)d * 21;
#pragma unroll
  for (int c = 0; c < 5; ++c) atomicAddF(op + c, np[c]);
#pragma unroll
  for (int j = 0; j < 16; ++j) atomicAddF(op + 5 + j, emb[j]);
}

// ---------------------------------------------------------------------------
// Kernel 2: per-node pre = (1+eps1)*(nodes@Wego) + agg1 ; x1 = pre@W1a + b1a.
// K=21 is too skinny for WMMA; f32 VALU, weights staged in LDS.
// ---------------------------------------------------------------------------
__global__ void k_node1_pre(const float* __restrict__ nodes, const float* __restrict__ agg1,
                            const float* __restrict__ Wego, const float* __restrict__ W1a,
                            const float* __restrict__ b1a, const float* __restrict__ eps1p,
                            float* __restrict__ x1, int N) {
  __shared__ float sWego[105], sW1a[1344], sb1a[64];
  for (int i = threadIdx.x; i < 105; i += blockDim.x) sWego[i] = Wego[i];
  for (int i = threadIdx.x; i < 1344; i += blockDim.x) sW1a[i] = W1a[i];
  for (int i = threadIdx.x; i < 64; i += blockDim.x) sb1a[i] = b1a[i];
  __syncthreads();
  int n = blockIdx.x * blockDim.x + threadIdx.x;
  if (n >= N) return;
  float e1 = 1.0f + eps1p[0];
  float nd[5];
#pragma unroll
  for (int k = 0; k < 5; ++k) nd[k] = nodes[(size_t)n * 5 + k];
  float pre[21];
#pragma unroll
  for (int c = 0; c < 21; ++c) {
    float ego = 0.f;
#pragma unroll
    for (int k = 0; k < 5; ++k) ego += nd[k] * sWego[k * 21 + c];
    pre[c] = e1 * ego + agg1[(size_t)n * 21 + c];
  }
  float* xo = x1 + (size_t)n * 64;
#pragma unroll 4
  for (int j = 0; j < 64; ++j) {
    float a = sb1a[j];
#pragma unroll
    for (int k = 0; k < 21; ++k) a += pre[k] * sW1a[k * 64 + j];
    xo[j] = a;
  }
}

// ---------------------------------------------------------------------------
// Kernel 3: per-channel sum / sum-of-squares for BatchNorm (grid*block % C == 0
// so each thread stays on one channel; 2 atomics per thread at the end).
// ---------------------------------------------------------------------------
__global__ void k_stats(const float* __restrict__ x, float* __restrict__ sums,
                        float* __restrict__ sqs, long total, int C) {
  long stride = (long)gridDim.x * blockDim.x;
  long i0 = (long)blockIdx.x * blockDim.x + threadIdx.x;
  int c = (int)(i0 % C);
  float s = 0.f, q = 0.f;
  for (long i = i0; i < total; i += stride) { float v = x[i]; s += v; q += v * v; }
  atomicAddF(sums + c, s);
  atomicAddF(sqs + c, q);
}

// ---------------------------------------------------------------------------
// Kernel 4: layer-1 post: h = GELU(BN(x1)) @ W1b + b1b  (WMMA, [16x64]x[64x64]
// per wave). Writes h (f32, for (1+eps2)*h) and hHalf (f16, for edge gather).
// ---------------------------------------------------------------------------
__global__ void k_post1(const float* __restrict__ x1,
                        const float* __restrict__ sums, const float* __restrict__ sqs,
                        const float* __restrict__ g, const float* __restrict__ beta,
                        const float* __restrict__ W1b, const float* __restrict__ b1b,
                        float* __restrict__ h, _Float16* __restrict__ hHalf, int N) {
  __shared__ _Float16 sW[64 * 64];           // swizzled B fragments
  __shared__ float sScale[64], sShift[64], sBias[64];
  __shared__ _Float16 aT[8][16][72];
  stage_w_swizzled<64, 64>(W1b, sW);
  if (threadIdx.x < 64) {
    int c = threadIdx.x;
    float mean = sums[c] / (float)N;
    float var  = sqs[c] / (float)N - mean * mean;
    float sc   = g[c] * rsqrtf(var + 1e-5f);
    sScale[c] = sc;
    sShift[c] = beta[c] - mean * sc;
    sBias[c]  = b1b[c];
  }
  __syncthreads();
  int wave = threadIdx.x >> 5, lane = threadIdx.x & 31;
  int rowBase = blockIdx.x * 128 + wave * 16;
  bool active = rowBase < N;  // 16 | N: active waves are fully in range
  if (active) {
    for (int idx = lane; idx < 16 * 64; idx += 32) {
      int r = idx >> 6, c = idx & 63;
      float v = x1[(size_t)(rowBase + r) * 64 + c];
      aT[wave][r][c] = (_Float16)gelu_exact(v * sScale[c] + sShift[c]);
    }
  }
  __syncthreads();
  if (!active) return;
  int col = lane & 15, kg = lane >> 4;
#pragma unroll
  for (int nt = 0; nt < 4; ++nt) {
    float bias = sBias[nt * 16 + col];
    v8f acc;
#pragma unroll
    for (int i = 0; i < 8; ++i) acc[i] = bias;
#pragma unroll
    for (int kt = 0; kt < 2; ++kt) {
      v16h a = load_a_frag(&aT[wave][0][0], lane, kt, 72);
      v16h b = load_b_frag_sw(sW, lane, kt, 4, nt);
      acc = __builtin_amdgcn_wmma_f32_16x16x32_f16(false, a, false, b, (short)0, acc,
                                                   false, false);
    }
#pragma unroll
    for (int v = 0; v < 8; ++v) {
      size_t o = (size_t)(rowBase + v + kg * 8) * 64 + nt * 16 + col;
      h[o] = acc[v];
      hHalf[o] = (_Float16)acc[v];
    }
  }
}

// ---------------------------------------------------------------------------
// Kernel 5: edge gather/scatter for layer 2: agg2[dst] += h[src] (64 chans).
// 8 threads/edge, 16B f16 vector gathers, f32 atomic scatter.
// ---------------------------------------------------------------------------
__global__ void k_edge2(const int* __restrict__ eidx, const _Float16* __restrict__ hHalf,
                        float* __restrict__ agg2, int E) {
  long tid = (long)blockIdx.x * blockDim.x + threadIdx.x;
  long e = tid >> 3;
  if (e >= E) return;
  int cb = ((int)tid & 7) * 8;
  int s = eidx[e], d = eidx[(size_t)E + e];
  union { uint4 u; _Float16 hh[8]; } buf;
  buf.u = *reinterpret_cast<const uint4*>(hHalf + (size_t)s * 64 + cb);
  float* op = agg2 + (size_t)d * 64 + cb;
#pragma unroll
  for (int i = 0; i < 8; ++i) atomicAddF(op + i, (float)buf.hh[i]);
}

// ---------------------------------------------------------------------------
// Kernel 6: layer-2 pre: x2 = ((1+eps2)*h + agg2) @ W2a + b2a  (WMMA,
// [16x64]x[64x128] per wave).
// ---------------------------------------------------------------------------
__global__ void k_pre2(const float* __restrict__ h, const float* __restrict__ agg2,
                       const float* __restrict__ W2a, const float* __restrict__ b2a,
                       const float* __restrict__ eps2p, float* __restrict__ x2, int N) {
  __shared__ _Float16 sW[64 * 128];          // swizzled B fragments
  __shared__ float sBias[128];
  __shared__ _Float16 aT[8][16][72];
  stage_w_swizzled<64, 128>(W2a, sW);
  if (threadIdx.x < 128) sBias[threadIdx.x] = b2a[threadIdx.x];
  __syncthreads();
  float e2 = 1.0f + eps2p[0];
  int wave = threadIdx.x >> 5, lane = threadIdx.x & 31;
  int rowBase = blockIdx.x * 128 + wave * 16;
  bool active = rowBase < N;
  if (active) {
    for (int idx = lane; idx < 16 * 64; idx += 32) {
      int r = idx >> 6, c = idx & 63;
      size_t o = (size_t)(rowBase + r) * 64 + c;
      aT[wave][r][c] = (_Float16)(e2 * h[o] + agg2[o]);
    }
  }
  __syncthreads();
  if (!active) return;
  int col = lane & 15, kg = lane >> 4;
#pragma unroll
  for (int nt = 0; nt < 8; ++nt) {
    float bias = sBias[nt * 16 + col];
    v8f acc;
#pragma unroll
    for (int i = 0; i < 8; ++i) acc[i] = bias;
#pragma unroll
    for (int kt = 0; kt < 2; ++kt) {
      v16h a = load_a_frag(&aT[wave][0][0], lane, kt, 72);
      v16h b = load_b_frag_sw(sW, lane, kt, 8, nt);
      acc = __builtin_amdgcn_wmma_f32_16x16x32_f16(false, a, false, b, (short)0, acc,
                                                   false, false);
    }
#pragma unroll
    for (int v = 0; v < 8; ++v)
      x2[(size_t)(rowBase + v + kg * 8) * 128 + nt * 16 + col] = acc[v];
  }
}

// ---------------------------------------------------------------------------
// Kernel 7: layer-2 post + pooling: h2 = GELU(BN(x2)) @ W2b + b2b; then
// pooled[batch_idx[row]] += h2 row (f32 atomics).  WMMA [16x128]x[128x128].
// Block = 128 threads (4 waves) to keep static LDS ~51 KB.
// ---------------------------------------------------------------------------
__global__ void k_post2_pool(const float* __restrict__ x2,
                             const float* __restrict__ sums, const float* __restrict__ sqs,
                             const float* __restrict__ g, const float* __restrict__ beta,
                             const float* __restrict__ W2b, const float* __restrict__ b2b,
                             const int* __restrict__ bidx,
                             float* __restrict__ pooled, int N) {
  __shared__ _Float16 sW[128 * 128];  // 32 KB, swizzled B fragments
  __shared__ float sScale[128], sShift[128], sBias[128];
  __shared__ _Float16 aT[4][16][136];  // ~17 KB
  stage_w_swizzled<128, 128>(W2b, sW);
  if (threadIdx.x < 128) {
    int c = threadIdx.x;
    float mean = sums[c] / (float)N;
    float var  = sqs[c] / (float)N - mean * mean;
    float sc   = g[c] * rsqrtf(var + 1e-5f);
    sScale[c] = sc;
    sShift[c] = beta[c] - mean * sc;
    sBias[c]  = b2b[c];
  }
  __syncthreads();
  int wave = threadIdx.x >> 5, lane = threadIdx.x & 31;
  int rowBase = blockIdx.x * 64 + wave * 16;
  bool active = rowBase < N;  // 16 | N
  if (active) {
    for (int idx = lane; idx < 16 * 128; idx += 32) {
      int r = idx >> 7, c = idx & 127;
      float v = x2[(size_t)(rowBase + r) * 128 + c];
      aT[wave][r][c] = (_Float16)gelu_exact(v * sScale[c] + sShift[c]);
    }
  }
  __syncthreads();
  if (!active) return;
  int col = lane & 15, kg = lane >> 4;
#pragma unroll
  for (int nt = 0; nt < 8; ++nt) {
    float bias = sBias[nt * 16 + col];
    v8f acc;
#pragma unroll
    for (int i = 0; i < 8; ++i) acc[i] = bias;
#pragma unroll
    for (int kt = 0; kt < 4; ++kt) {
      v16h a = load_a_frag(&aT[wave][0][0], lane, kt, 136);
      v16h b = load_b_frag_sw(sW, lane, kt, 8, nt);
      acc = __builtin_amdgcn_wmma_f32_16x16x32_f16(false, a, false, b, (short)0, acc,
                                                   false, false);
    }
#pragma unroll
    for (int v = 0; v < 8; ++v) {
      int r = rowBase + v + kg * 8;
      atomicAddF(pooled + (size_t)bidx[r] * 128 + nt * 16 + col, acc[v]);
    }
  }
}

// ---------------------------------------------------------------------------
// Kernel 8: out = GELU(pooled @ Wp + bp)   (64x128 output — trivial, f32 VALU)
// ---------------------------------------------------------------------------
__global__ void k_final(const float* __restrict__ pooled, const float* __restrict__ Wp,
                        const float* __restrict__ bp, float* __restrict__ out, int total) {
  int t = blockIdx.x * blockDim.x + threadIdx.x;
  if (t >= total) return;
  int r = t >> 7, c = t & 127;
  const float* pr = pooled + (size_t)r * 128;
  float a = bp[c];
#pragma unroll 8
  for (int k = 0; k < 128; ++k) a += pr[k] * Wp[k * 128 + c];
  out[t] = gelu_exact(a);
}

// ---------------------------------------------------------------------------
extern "C" void kernel_launch(void* const* d_in, const int* in_sizes, int n_in,
                              void* d_out, int out_size, void* d_ws, size_t ws_size,
                              hipStream_t stream) {
  const float* nodes = (const float*)d_in[0];
  const int*   eidx  = (const int*)d_in[1];
  const float* efeat = (const float*)d_in[2];
  const int*   bidx  = (const int*)d_in[3];
  // d_in[4] is B (device scalar); B derived from out_size below.
  const float* We1 = (const float*)d_in[5];
  const float* be1 = (const float*)d_in[6];
  const float* We2 = (const float*)d_in[7];
  const float* be2 = (const float*)d_in[8];
  const float* Wego = (const float*)d_in[9];
  const float* W1a = (const float*)d_in[10];
  const float* b1a = (const float*)d_in[11];
  const float* g1  = (const float*)d_in[12];
  const float* bt1 = (const float*)d_in[13];
  const float* W1b = (const float*)d_in[14];
  const float* b1b = (const float*)d_in[15];
  const float* eps1 = (const float*)d_in[16];
  const float* W2a = (const float*)d_in[17];
  const float* b2a = (const float*)d_in[18];
  const float* g2  = (const float*)d_in[19];
  const float* bt2 = (const float*)d_in[20];
  const float* W2b = (const float*)d_in[21];
  const float* b2b = (const float*)d_in[22];
  const float* eps2 = (const float*)d_in[23];
  const float* Wp  = (const float*)d_in[24];
  const float* bp  = (const float*)d_in[25];

  const int N = in_sizes[0] / 5;
  const int E = in_sizes[1] / 2;
  const int B = out_size / 128;

  // ---- workspace carve-out (~285 MB) ----
  char* ws = (char*)d_ws;
  size_t off = 0;
  auto carve = [&](size_t bytes) -> void* {
    void* p = ws + off;
    off = (off + bytes + 255) & ~(size_t)255;
    return p;
  };
  float*    agg1   = (float*)carve((size_t)N * 21 * 4);
  float*    x1     = (float*)carve((size_t)N * 64 * 4);
  float*    h      = (float*)carve((size_t)N * 64 * 4);
  _Float16* hHalf  = (_Float16*)carve((size_t)N * 64 * 2);
  float*    agg2   = (float*)carve((size_t)N * 64 * 4);
  float*    x2     = (float*)carve((size_t)N * 128 * 4);
  float*    pooled = (float*)carve((size_t)B * 128 * 4);
  float*    sums1  = (float*)carve(64 * 4);
  float*    sqs1   = (float*)carve(64 * 4);
  float*    sums2  = (float*)carve(128 * 4);
  float*    sqs2   = (float*)carve(128 * 4);

  hipMemsetAsync(agg1, 0, (size_t)N * 21 * 4, stream);
  hipMemsetAsync(agg2, 0, (size_t)N * 64 * 4, stream);
  hipMemsetAsync(pooled, 0, (size_t)B * 128 * 4, stream);
  hipMemsetAsync(sums1, 0, 64 * 4, stream);
  hipMemsetAsync(sqs1, 0, 64 * 4, stream);
  hipMemsetAsync(sums2, 0, 128 * 4, stream);
  hipMemsetAsync(sqs2, 0, 128 * 4, stream);

  k_edge1<<<(E + 255) / 256, 256, 0, stream>>>(nodes, eidx, efeat, We1, be1, We2, be2,
                                               agg1, E);
  k_node1_pre<<<(N + 255) / 256, 256, 0, stream>>>(nodes, agg1, Wego, W1a, b1a, eps1,
                                                   x1, N);
  k_stats<<<512, 256, 0, stream>>>(x1, sums1, sqs1, (long)N * 64, 64);
  k_post1<<<(N + 127) / 128, 256, 0, stream>>>(x1, sums1, sqs1, g1, bt1, W1b, b1b,
                                               h, hHalf, N);
  k_edge2<<<(int)(((long)E * 8 + 255) / 256), 256, 0, stream>>>(eidx, hHalf, agg2, E);
  k_pre2<<<(N + 127) / 128, 256, 0, stream>>>(h, agg2, W2a, b2a, eps2, x2, N);
  k_stats<<<512, 256, 0, stream>>>(x2, sums2, sqs2, (long)N * 128, 128);
  k_post2_pool<<<(N + 63) / 64, 128, 0, stream>>>(x2, sums2, sqs2, g2, bt2, W2b, b2b,
                                                  bidx, pooled, N);
  k_final<<<(B * 128 + 255) / 256, 256, 0, stream>>>(pooled, Wp, bp, (float*)d_out,
                                                     B * 128);
}
